// Attention_3633542333119
// MI455X (gfx1250) — matrix operand order repaired and checked by reading, not compiled, not run on hardware
//
#include <hip/hip_runtime.h>
#include <hip/hip_bf16.h>

typedef __attribute__((ext_vector_type(8)))  float        v8f;
typedef __attribute__((ext_vector_type(16))) __bf16       v16bf;
typedef __attribute__((ext_vector_type(4)))  unsigned int v4u;
typedef __attribute__((ext_vector_type(2)))  unsigned int v2u;

#define H_DIM 512
#define S_DIM 2048
#define B_DIM 128

// round-to-nearest-even fp32 -> bf16 (bit trick)
__device__ __forceinline__ unsigned short f2bf(float f) {
    unsigned u = __float_as_uint(f);
    unsigned r = u + 0x7FFFu + ((u >> 16) & 1u);
    return (unsigned short)(r >> 16);
}
__device__ __forceinline__ float bf2f(unsigned short h) {
    return __uint_as_float(((unsigned)h) << 16);
}

union BFrag { v16bf bf; v4u u[2]; };

// generic pointer to __shared__ -> 32-bit LDS byte offset (low 32 bits of
// the LDS-aperture generic address)
__device__ __forceinline__ unsigned lds_off(const void* p) {
    return (unsigned)(unsigned long long)(uintptr_t)p;
}

// gfx1250 async global -> LDS copy, 16 bytes per lane (ASYNCcnt tracked)
__device__ __forceinline__ void async_g2l_b128(unsigned lds_addr, const void* gaddr) {
    asm volatile("global_load_async_to_lds_b128 %0, %1, off"
                 :: "v"(lds_addr), "v"(gaddr)
                 : "memory");
}

// ---------------------------------------------------------------------------
// Kernel 1: q_proj[b][i] = sum_k q[b][k] * Wq[i][k]   (tiny: 65536 threads)
// ---------------------------------------------------------------------------
__global__ void qproj_kernel(const float* __restrict__ q,
                             const float* __restrict__ Wq,
                             float* __restrict__ qp) {
    int idx = blockIdx.x * blockDim.x + threadIdx.x;   // 0 .. 65535
    int b = idx >> 9;
    int i = idx & (H_DIM - 1);
    const float* qr = q  + (size_t)b * H_DIM;
    const float* wr = Wq + (size_t)i * H_DIM;
    float acc = 0.f;
#pragma unroll 8
    for (int k = 0; k < H_DIM; ++k) acc = fmaf(qr[k], wr[k], acc);
    qp[idx] = acc;
}

// ---------------------------------------------------------------------------
// Kernel 2: split Wref (fp32) into bf16 hi/lo planes in workspace
// ---------------------------------------------------------------------------
__global__ void wconv_kernel(const float* __restrict__ Wref,
                             unsigned short* __restrict__ bh,
                             unsigned short* __restrict__ bl) {
    int idx = blockIdx.x * blockDim.x + threadIdx.x;   // 0 .. 262143
    float f = Wref[idx];
    unsigned short h = f2bf(f);
    bh[idx] = h;
    bl[idx] = f2bf(f - bf2f(h));
}

// ---------------------------------------------------------------------------
// Kernel 3: fused GEMM (bf16x3 WMMA, async-LDS double-buffered B) + epilogue
//   grid: 4096 workgroups of 512 threads (16 waves, 4x4 wave grid)
//   each workgroup: 64 rows x 512 cols of ref_proj, plus u[64]
// ---------------------------------------------------------------------------
__global__ void __launch_bounds__(512)
bahdanau_gemm_kernel(const float* __restrict__ ref,
                     const float* __restrict__ vvec,
                     const float* __restrict__ qp_ws,
                     const unsigned short* __restrict__ bh_ws,
                     const unsigned short* __restrict__ bl_ws,
                     float* __restrict__ u_out,
                     float* __restrict__ refp_out) {
    __shared__ unsigned short Ah[64 * 32];
    __shared__ unsigned short Al[64 * 32];
    __shared__ unsigned short Bh[2][H_DIM * 32];   // double-buffered (async fill)
    __shared__ unsigned short Bl[2][H_DIM * 32];
    __shared__ float qp_s[H_DIM];
    __shared__ float v_s[H_DIM];
    __shared__ float u_s[64];

    const int  tid    = threadIdx.x;          // 0..511
    const int  lane   = tid & 31;
    const int  wid    = tid >> 5;             // 0..15
    const int  wave_m = wid >> 2;             // 0..3  (16 rows each)
    const int  wave_n = wid & 3;              // 0..3  (128 cols each)
    const int  half   = lane >> 4;            // 0 | 1
    const int  l15    = lane & 15;
    const long m0     = (long)blockIdx.x * 64;
    const int  b      = (int)(m0 >> 11);      // m0 / 2048

    v8f acc[8];
#pragma unroll
    for (int t = 0; t < 8; ++t)
#pragma unroll
        for (int i = 0; i < 8; ++i) acc[t][i] = 0.f;

    const int a_row = tid >> 3;               // 0..63
    const int a_kk  = (tid & 7) << 2;         // 0,4,..,28

    // async-issue one 512x32 bf16 hi/lo B slab into buffer `dstbuf`
    auto issue_b_slab = [&](int k0, int dstbuf) {
        const char* sh = (const char*)(bh_ws + (size_t)tid * H_DIM + k0);
        const char* sl = (const char*)(bl_ws + (size_t)tid * H_DIM + k0);
        unsigned dh = lds_off(&Bh[dstbuf][tid * 32]);
        unsigned dl = lds_off(&Bl[dstbuf][tid * 32]);
#pragma unroll
        for (int c = 0; c < 4; ++c) {          // 4 x 16B = full 64B row slice
            async_g2l_b128(dh + 16 * c, sh + 16 * c);
            async_g2l_b128(dl + 16 * c, sl + 16 * c);
        }
    };

    // one K step: fill A slab, prefetch next B slab, run 24 WMMAs
    // `buf` is always a literal at the call site -> constant LDS offsets
    auto k_step = [&](int k0, int buf) {
        __syncthreads();   // previous iteration's LDS consumers are done

        // ---- fill A slab: 64 rows x 32 k, fp32 -> bf16 hi/lo (VALU path) ----
        {
            const float4 f4 =
                *(const float4*)(ref + (m0 + a_row) * H_DIM + k0 + a_kk);
            unsigned short h0 = f2bf(f4.x), h1 = f2bf(f4.y);
            unsigned short h2 = f2bf(f4.z), h3 = f2bf(f4.w);
            unsigned short l0 = f2bf(f4.x - bf2f(h0));
            unsigned short l1 = f2bf(f4.y - bf2f(h1));
            unsigned short l2 = f2bf(f4.z - bf2f(h2));
            unsigned short l3 = f2bf(f4.w - bf2f(h3));
            v2u ph, pl;
            ph.x = (unsigned)h0 | ((unsigned)h1 << 16);
            ph.y = (unsigned)h2 | ((unsigned)h3 << 16);
            pl.x = (unsigned)l0 | ((unsigned)l1 << 16);
            pl.y = (unsigned)l2 | ((unsigned)l3 << 16);
            *(v2u*)&Ah[a_row * 32 + a_kk] = ph;
            *(v2u*)&Al[a_row * 32 + a_kk] = pl;
            if (k0 + 32 < H_DIM)  // gfx1250 global_prefetch_b8 for next slab
                __builtin_prefetch(ref + (m0 + a_row) * H_DIM + k0 + 32 + a_kk, 0, 0);
        }

        // ---- async-issue NEXT B slab into the other buffer, wait for this one
        if (k0 + 32 < H_DIM) {
            issue_b_slab(k0 + 32, buf ^ 1);
            // 8 outstanding from slab k+1; slab k (issued earlier, in-order)
            // is complete once ASYNCcnt <= 8
            asm volatile("s_wait_asynccnt 0x8" ::: "memory");
        } else {
            asm volatile("s_wait_asynccnt 0x0" ::: "memory");
        }
        __syncthreads();

        // ---- load A fragments (ISA 16-bit A 16x32 layout) ----
        const int am = wave_m * 16 + l15;
        const unsigned short* arh = &Ah[am * 32];
        const unsigned short* arl = &Al[am * 32];
        BFrag ah, al;
        ah.u[0] = *(const v4u*)&arh[half * 8];
        ah.u[1] = *(const v4u*)&arh[16 + half * 8];
        al.u[0] = *(const v4u*)&arl[half * 8];
        al.u[1] = *(const v4u*)&arl[16 + half * 8];

        // ---- 8 N-tiles: bf16x3 WMMA accumulation ----
#pragma unroll
        for (int t = 0; t < 8; ++t) {
            const int bn = wave_n * 128 + t * 16 + l15;
            const unsigned short* brh = &Bh[buf][bn * 32 + half * 8];
            const unsigned short* brl = &Bl[buf][bn * 32 + half * 8];
            BFrag bhf, blf;
            bhf.u[0] = *(const v4u*)&brh[0];
            bhf.u[1] = *(const v4u*)&brh[16];
            blf.u[0] = *(const v4u*)&brl[0];
            blf.u[1] = *(const v4u*)&brl[16];

            acc[t] = __builtin_amdgcn_wmma_f32_16x16x32_bf16(
                false, al.bf, false, bhf.bf, (short)0, acc[t], false, false);
            acc[t] = __builtin_amdgcn_wmma_f32_16x16x32_bf16(
                false, ah.bf, false, blf.bf, (short)0, acc[t], false, false);
            acc[t] = __builtin_amdgcn_wmma_f32_16x16x32_bf16(
                false, ah.bf, false, bhf.bf, (short)0, acc[t], false, false);
        }
    };

    // ---- prologue: kick off async B slab 0, stage per-workgroup vectors ----
    issue_b_slab(0, 0);
    qp_s[tid] = qp_ws[(size_t)b * H_DIM + tid];
    v_s[tid]  = vvec[tid];
    if (tid < 64) u_s[tid] = 0.f;

    // ---- K loop, unrolled x2 so the LDS buffer parity is compile-time ----
    for (int kk = 0; kk < H_DIM; kk += 64) {
        k_step(kk, 0);
        k_step(kk + 32, 1);
    }

    // ---- epilogue: store ref_proj, fused tanh(qp + rp) . v reduction ----
#pragma unroll
    for (int t = 0; t < 8; ++t) {
        const int   hcol = wave_n * 128 + t * 16 + l15;
        const float qph  = qp_s[hcol];
        const float vh   = v_s[hcol];
#pragma unroll
        for (int r = 0; r < 8; ++r) {
            const int   m   = wave_m * 16 + r + half * 8;
            const float val = acc[t][r];
            refp_out[(m0 + m) * (long)H_DIM + hcol] = val;
            float c = tanhf(val + qph) * vh;
            c += __shfl_xor(c, 1, 32);
            c += __shfl_xor(c, 2, 32);
            c += __shfl_xor(c, 4, 32);
            c += __shfl_xor(c, 8, 32);
            if (l15 == 0) atomicAdd(&u_s[m], c);   // ds_add_f32
        }
    }
    __syncthreads();
    if (tid < 64) u_out[m0 + tid] = u_s[tid];
}

// ---------------------------------------------------------------------------
extern "C" void kernel_launch(void* const* d_in, const int* in_sizes, int n_in,
                              void* d_out, int out_size, void* d_ws, size_t ws_size,
                              hipStream_t stream) {
    const float* q    = (const float*)d_in[0];   // [128, 512]
    const float* ref  = (const float*)d_in[1];   // [262144, 512]
    const float* vvec = (const float*)d_in[2];   // [512]
    const float* Wq   = (const float*)d_in[3];   // [512, 512]
    const float* Wref = (const float*)d_in[4];   // [512, 512]

    float* out      = (float*)d_out;
    float* u_out    = out;                                   // [128*2048]
    float* refp_out = out + (size_t)B_DIM * S_DIM;           // [128*2048*512]

    // workspace: q_proj (256KB) | Wref_hi bf16 (512KB) | Wref_lo bf16 (512KB)
    float*          qp_ws = (float*)d_ws;
    unsigned short* bh_ws = (unsigned short*)((char*)d_ws + (size_t)B_DIM * H_DIM * sizeof(float));
    unsigned short* bl_ws = bh_ws + (size_t)H_DIM * H_DIM;

    hipLaunchKernelGGL(qproj_kernel, dim3((B_DIM * H_DIM) / 256), dim3(256), 0, stream,
                       q, Wq, qp_ws);
    hipLaunchKernelGGL(wconv_kernel, dim3((H_DIM * H_DIM) / 256), dim3(256), 0, stream,
                       Wref, bh_ws, bl_ws);

    const int mblocks = (B_DIM * S_DIM) / 64;   // 4096
    hipLaunchKernelGGL(bahdanau_gemm_kernel, dim3(mblocks), dim3(512), 0, stream,
                       ref, vvec, qp_ws, bh_ws, bl_ws, u_out, refp_out);
}
